// ZeroShotCosineModel_53532472377392
// MI455X (gfx1250) — compile-verified, matrix operand order checked by reading, not run on hardware
//
#include <hip/hip_runtime.h>
#include <math.h>

#define B_    8
#define H_    128
#define W_    128
#define HW_   (H_ * W_)
#define E_    1024
#define K_    16
#define TOPK_ 4

typedef __attribute__((ext_vector_type(16))) _Float16 v16h;
typedef __attribute__((ext_vector_type(8)))  _Float16 v8h;
typedef __attribute__((ext_vector_type(8)))  float    v8f;
typedef __attribute__((ext_vector_type(4)))  float    f4;

// ---------------------------------------------------------------------------
// Kernel 1: L2 norms of descriptors (B*K rows) and queries (B rows).
// ---------------------------------------------------------------------------
__global__ __launch_bounds__(256) void zs_norms_kernel(
    const float* __restrict__ desc, const float* __restrict__ query,
    float* __restrict__ dnorm, float* __restrict__ qnorm) {
  int idx = blockIdx.x;
  int tid = threadIdx.x;
  const float* src = (idx < B_ * K_) ? (desc + (size_t)idx * E_)
                                     : (query + (size_t)(idx - B_ * K_) * E_);
  float s = 0.f;
  for (int e = tid; e < E_; e += 256) { float x = src[e]; s += x * x; }
  __shared__ float red[256];
  red[tid] = s;
  __syncthreads();
  for (int off = 128; off > 0; off >>= 1) {
    if (tid < off) red[tid] += red[tid + off];
    __syncthreads();
  }
  if (tid == 0) {
    float n = sqrtf(red[0]);
    if (idx < B_ * K_) dnorm[idx] = n;
    else qnorm[idx - B_ * K_] = n;
  }
}

// ---------------------------------------------------------------------------
// Kernel 2: fused score GEMM (WMMA f16 -> f32 acc) + per-pixel feature norms.
// One wave computes a 16(M=pixels) x 16(N=descriptors) tile over E=1024.
// Block = 8 waves = 128 pixels; grid = (HW/128, B).
// Descriptors staged to LDS as f16 with padded rows (bank-conflict stagger).
// K loop unrolled 4x: 16 global_load_b128 + 8 ds_load_b128 + 4 wmma per trip.
// ---------------------------------------------------------------------------
__global__ __launch_bounds__(256) void zs_score_kernel(
    const float* __restrict__ feat, const float* __restrict__ desc,
    const float* __restrict__ dnorm, float* __restrict__ scores,
    float* __restrict__ fnorm) {
  constexpr int EP = E_ + 8;  // padded LDS row stride (halfs)
  __shared__ _Float16 sdesc[K_ * EP];
  __shared__ float sdn[K_];

  const int tid = threadIdx.x;
  const int b = blockIdx.y;

  // Stage this batch's 16x1024 descriptor block into LDS as f16.
  const float* dsc = desc + (size_t)b * K_ * E_;
  for (int i = tid; i < K_ * E_; i += 256) {
    int k = i >> 10, e = i & (E_ - 1);
    sdesc[k * EP + e] = (_Float16)dsc[i];
  }
  if (tid < K_) sdn[tid] = dnorm[b * K_ + tid];
  __syncthreads();

  const int wave = tid >> 5, lane = tid & 31;
  const int hi = lane >> 4, lm = lane & 15;
  const int pixbase = blockIdx.x * 128 + wave * 16;
  const int row = pixbase + lm;

  const float* rowp = feat + ((size_t)b * HW_ + row) * (size_t)E_;
  const _Float16* drow = sdesc + lm * EP;

  v8f c = {};
  float ssq = 0.f;

#pragma unroll 4
  for (int kb = 0; kb < E_; kb += 32) {
    // A fragment (16-bit, 16x32): lanes 0-15 hold K kb+{0..7,16..23},
    // lanes 16-31 hold K kb+{8..15,24..31}; M = lane%16.
    const int e0 = kb + hi * 8;
    const int e1 = kb + 16 + hi * 8;
    f4 x0 = *(const f4*)(rowp + e0);
    f4 x1 = *(const f4*)(rowp + e0 + 4);
    f4 x2 = *(const f4*)(rowp + e1);
    f4 x3 = *(const f4*)(rowp + e1 + 4);

    // Speculative streaming prefetch one unroll-group ahead (dropped if
    // translation fails; overshoot lands in the next feature row).
    __builtin_prefetch(rowp + kb + 128, 0, 1);

    v16h av, bv;
#pragma unroll
    for (int j = 0; j < 4; j++) {
      av[j]      = (_Float16)x0[j];
      av[4 + j]  = (_Float16)x1[j];
      av[8 + j]  = (_Float16)x2[j];
      av[12 + j] = (_Float16)x3[j];
      ssq += x0[j] * x0[j] + x1[j] * x1[j] + x2[j] * x2[j] + x3[j] * x3[j];
    }

    // B fragment (32x16): lane holds column n = lane%16, same K striping as A.
    v8h b0 = *(const v8h*)(drow + e0);
    v8h b1 = *(const v8h*)(drow + e1);
#pragma unroll
    for (int j = 0; j < 8; j++) { bv[j] = b0[j]; bv[8 + j] = b1[j]; }

    c = __builtin_amdgcn_wmma_f32_16x16x32_f16(
        /*neg_a=*/false, av, /*neg_b=*/false, bv,
        /*c_mod=*/(short)0, c, /*reuse_a=*/false, /*reuse_b=*/false);
  }

  // Full sum-of-squares for row lm: lanes l and l+16 cover complementary K.
  float ssqf = ssq + __shfl_xor(ssq, 16);
  float fr = sqrtf(ssqf);
  if (lane < 16) fnorm[(size_t)b * HW_ + row] = fr;

  // C layout: VGPR v holds (M = v + 8*hi, N = lane%16).
  const int N = lm;
  const float dn = fmaxf(sdn[N], 1e-12f);
#pragma unroll
  for (int v = 0; v < 8; v++) {
    int M = v + hi * 8;
    float fn = fmaxf(__shfl(fr, M), 1e-12f);
    scores[((size_t)(b * K_ + N)) * HW_ + pixbase + M] = c[v] / (fn * dn);
  }
}

// ---------------------------------------------------------------------------
// Kernel 3: greedy radius NMS, one block per (b,k). 4 rounds of suppressed
// argmax with first-index tie-break (== stable argsort-scan greedy).
// ---------------------------------------------------------------------------
__global__ __launch_bounds__(256) void zs_nms_kernel(
    const float* __restrict__ scores, const int* __restrict__ radp,
    int* __restrict__ rows, int* __restrict__ cols, int* __restrict__ valid) {
  const int blk = blockIdx.x;  // b*K + k
  const int tid = threadIdx.x;
  const int rad = *radp;
  const float* s = scores + (size_t)blk * HW_;

  __shared__ int ar[TOPK_], ac[TOPK_];
  __shared__ float rv[256];
  __shared__ int ri[256];

  for (int round = 0; round < TOPK_; round++) {
    float bv = -INFINITY;
    int bi = 0x7fffffff;
    for (int i = tid; i < HW_; i += 256) {
      int r = i >> 7, cc = i & (W_ - 1);
      bool sup = false;
      for (int t = 0; t < round; t++) {
        int dr = r - ar[t];  if (dr < 0) dr = -dr;
        int dc = cc - ac[t]; if (dc < 0) dc = -dc;
        sup = sup || (dr <= rad && dc <= rad);
      }
      float v = s[i];
      if (!sup && (v > bv || (v == bv && i < bi))) { bv = v; bi = i; }
    }
    rv[tid] = bv; ri[tid] = bi;
    __syncthreads();
    for (int off = 128; off > 0; off >>= 1) {
      if (tid < off) {
        float v2 = rv[tid + off]; int i2 = ri[tid + off];
        if (v2 > rv[tid] || (v2 == rv[tid] && i2 < ri[tid])) {
          rv[tid] = v2; ri[tid] = i2;
        }
      }
      __syncthreads();
    }
    if (tid == 0) { ar[round] = ri[0] >> 7; ac[round] = ri[0] & (W_ - 1); }
    __syncthreads();
  }
  if (tid < TOPK_) {
    rows[blk * TOPK_ + tid] = ar[tid];
    cols[blk * TOPK_ + tid] = ac[tid];
    valid[blk * TOPK_ + tid] = 1;
  }
}

// ---------------------------------------------------------------------------
// Kernel 4: per-batch value map + argmax. One block per batch.
// vmap (in ws) is zeroed, masked pixels get cos(feat[p], query), then
// first-index argmax over the full map (reproduces background==0 semantics).
// ---------------------------------------------------------------------------
__global__ __launch_bounds__(256) void zs_final_kernel(
    const float* __restrict__ feat, const float* __restrict__ query,
    const float* __restrict__ fnorm, const float* __restrict__ qnorm,
    const int* __restrict__ rows, const int* __restrict__ cols,
    const int* __restrict__ valid, const int* __restrict__ nbp,
    float* __restrict__ vmap, float* __restrict__ out) {
  const int b = blockIdx.x, tid = threadIdx.x;
  float* vm = vmap + (size_t)b * HW_;
  for (int i = tid; i < HW_; i += 256) vm[i] = 0.f;
  __threadfence_block();
  __syncthreads();

  const int nb = *nbp;
  const int side = 2 * nb + 1;
  const int cells = side * side;
  const int jobs = K_ * TOPK_ * cells;
  const float qn = qnorm[b];
  const float* q = query + (size_t)b * E_;

  const int wave = tid >> 5, lane = tid & 31;
  for (int j = wave; j < jobs; j += 8) {
    int cell = j / cells, dd = j % cells;
    int k = cell / TOPK_, t = cell % TOPK_;
    int base = (b * K_ + k) * TOPK_ + t;
    if (!valid[base]) continue;  // wave-uniform
    int r  = rows[base] + dd / side - nb;
    int cc = cols[base] + dd % side - nb;
    r  = r  < 0 ? 0 : (r  > H_ - 1 ? H_ - 1 : r);
    cc = cc < 0 ? 0 : (cc > W_ - 1 ? W_ - 1 : cc);
    int p = r * W_ + cc;
    const float* fp = feat + ((size_t)b * HW_ + p) * (size_t)E_;
    float sdot = 0.f;
    for (int e = lane; e < E_; e += 32) sdot += fp[e] * q[e];
#pragma unroll
    for (int off = 16; off > 0; off >>= 1) sdot += __shfl_xor(sdot, off);
    if (lane == 0) {
      float denom = fmaxf(fnorm[(size_t)b * HW_ + p] * qn, 1e-8f);
      vm[p] = sdot / denom;  // duplicates write identical values
    }
  }
  __threadfence_block();
  __syncthreads();

  __shared__ float rv[256];
  __shared__ int ri[256];
  float bv = -INFINITY;
  int bi = 0x7fffffff;
  for (int i = tid; i < HW_; i += 256) {
    float v = vm[i];
    if (v > bv || (v == bv && i < bi)) { bv = v; bi = i; }
  }
  rv[tid] = bv; ri[tid] = bi;
  __syncthreads();
  for (int off = 128; off > 0; off >>= 1) {
    if (tid < off) {
      float v2 = rv[tid + off]; int i2 = ri[tid + off];
      if (v2 > rv[tid] || (v2 == rv[tid] && i2 < ri[tid])) {
        rv[tid] = v2; ri[tid] = i2;
      }
    }
    __syncthreads();
  }
  if (tid == 0) {
    out[b]      = (float)ri[0];  // max_index (cast to output dtype)
    out[B_ + b] = rv[0];         // max_value
  }
}

// ---------------------------------------------------------------------------
extern "C" void kernel_launch(void* const* d_in, const int* in_sizes, int n_in,
                              void* d_out, int out_size, void* d_ws,
                              size_t ws_size, hipStream_t stream) {
  (void)in_sizes; (void)n_in; (void)out_size; (void)ws_size;
  const float* feat  = (const float*)d_in[0];  // (8,128,128,1024)
  const float* query = (const float*)d_in[1];  // (8,1024)
  const float* desc  = (const float*)d_in[2];  // (8,16,1024)
  // d_in[3] = top_k (4, baked into shapes)
  const int* nbp  = (const int*)d_in[4];       // neighborhood
  const int* radp = (const int*)d_in[5];       // nms_radius
  float* out = (float*)d_out;                  // 16 floats: index[8] ++ value[8]

  // Workspace carve-out (~9.4 MB total).
  float* wsf    = (float*)d_ws;
  float* scores = wsf;                                   // B*K*HW
  float* fnorm  = scores + (size_t)B_ * K_ * HW_;        // B*HW
  float* dnorm  = fnorm + (size_t)B_ * HW_;              // B*K
  float* qnorm  = dnorm + B_ * K_;                       // B
  float* vmap   = qnorm + B_;                            // B*HW
  int* rowsA  = (int*)(vmap + (size_t)B_ * HW_);         // B*K*4
  int* colsA  = rowsA + B_ * K_ * TOPK_;
  int* validA = colsA + B_ * K_ * TOPK_;

  zs_norms_kernel<<<B_ * K_ + B_, 256, 0, stream>>>(desc, query, dnorm, qnorm);
  zs_score_kernel<<<dim3(HW_ / 128, B_), 256, 0, stream>>>(feat, desc, dnorm,
                                                           scores, fnorm);
  zs_nms_kernel<<<B_ * K_, 256, 0, stream>>>(scores, radp, rowsA, colsA,
                                             validA);
  zs_final_kernel<<<B_, 256, 0, stream>>>(feat, query, fnorm, qnorm, rowsA,
                                          colsA, validA, nbp, vmap, out);
}
